// GroupedQueryAttention_63110249447831
// MI455X (gfx1250) — matrix operand order repaired and checked by reading, not compiled, not run on hardware
//
#include <hip/hip_runtime.h>

// ---------------- problem constants ----------------
constexpr int B_   = 2;
constexpr int T_   = 2048;
constexpr int E_   = 2048;
constexpr int H_   = 16;
constexpr int HKV_ = 4;
constexpr int D_   = 128;
constexpr int BT_  = B_ * T_;            // 4096 rows
constexpr float LOG_ROPE_BASE = 9.210340371976184f; // ln(10000)

typedef _Float16 v16h __attribute__((ext_vector_type(16)));
typedef _Float16 v8h  __attribute__((ext_vector_type(8)));
typedef float    v8f  __attribute__((ext_vector_type(8)));

#define CAT16(lo, hi) __builtin_shufflevector((lo), (hi), 0,1,2,3,4,5,6,7,8,9,10,11,12,13,14,15)

static __device__ __forceinline__ v8f wmma_f16(v16h a, v16h b, v8f c) {
  // D = A(16x32 f16) * B(32x16 f16) + C(16x16 f32)
  return __builtin_amdgcn_wmma_f32_16x16x32_f16(false, a, false, b, (short)0, c, false, false);
}

// ---- CDNA5 async copy: global -> LDS, 16 bytes/lane, tracked by ASYNCcnt ----
static __device__ __forceinline__ void async_copy_b128(const void* g, void* lds) {
  // generic pointer to __shared__ has the workgroup-relative LDS address in its
  // low 32 bits (AS3 value), which is what VDST of the async op consumes.
  unsigned loff = (unsigned)(unsigned long long)lds;
  unsigned long long ga = (unsigned long long)g;
  asm volatile("global_load_async_to_lds_b128 %0, %1, off"
               :: "v"(loff), "v"(ga)
               : "memory");
}
static __device__ __forceinline__ void wait_asynccnt0() {
  asm volatile("s_wait_asynccnt 0" ::: "memory");
}

// ---------------- elementwise f32 -> f16 ----------------
__global__ void cvt_f32_f16_kernel(const float* __restrict__ in, _Float16* __restrict__ out, int n) {
  int i = blockIdx.x * blockDim.x + threadIdx.x;
  if (i < n) out[i] = (_Float16)in[i];
}

// ---------------- RoPE (f32 in -> f16 out), nheads heads of D_=128 ----------------
__global__ void rope_kernel(const float* __restrict__ in, _Float16* __restrict__ out,
                            int nheads, int n) {
  int idx = blockIdx.x * blockDim.x + threadIdx.x;
  if (idx >= n) return;
  const int hp  = nheads * (D_ / 2);     // pairs per row
  int row = idx / hp;
  int rem = idx - row * hp;
  int hh  = rem >> 6;                    // D_/2 == 64
  int p   = rem & 63;
  int t   = row & (T_ - 1);              // T_ power of two
  float w   = __expf(-((2.0f * (float)p) / (float)D_) * LOG_ROPE_BASE);
  float ang = (float)t * w;
  float c = cosf(ang), sn = sinf(ang);
  size_t base = (size_t)row * (nheads * D_) + (size_t)hh * D_ + 2 * p;
  float re = in[base], im = in[base + 1];
  out[base]     = (_Float16)(re * c - im * sn);
  out[base + 1] = (_Float16)(re * sn + im * c);
}

// ---------------- f16 x f16 -> f32 GEMM, 128x64 block tile, async A staging ----
// A: MxK row-major, B: KxN row-major, C: MxN row-major f32. M%128==0, N%64==0, K%32==0.
__global__ __launch_bounds__(256) void gemm_f16_kernel(const _Float16* __restrict__ A,
                                                       const _Float16* __restrict__ Bm,
                                                       float* __restrict__ C,
                                                       int M, int N, int K) {
  __shared__ alignas(16) _Float16 ldsA[128][40];   // [m][k], padded
  __shared__ alignas(16) _Float16 ldsBt[64][40];   // [n][k], padded (transposed)

  const int tid  = threadIdx.x;
  const int lane = tid & 31;
  const int w    = tid >> 5;           // 8 waves, wave w owns rows w*16..w*16+15
  const int ln   = lane & 15;
  const int hl   = lane >> 4;          // half-wave id
  const int kb   = hl ? 8 : 0;         // A-fragment K base
  const int kbb  = hl ? 16 : 0;        // unused; B fragments also use kb
  const int mbase = blockIdx.y * 128;
  const int nbase = blockIdx.x * 64;

  // staging coordinates
  const int arow = tid >> 1, acol = (tid & 1) * 16;   // A: 128 rows x 32 k
  const int krow = tid >> 3, ncol = (tid & 7) * 8;    // B: 32 k x 64 n

  v8f acc[4];
#pragma unroll
  for (int i = 0; i < 4; ++i) acc[i] = (v8f){0,0,0,0,0,0,0,0};

  for (int k0 = 0; k0 < K; k0 += 32) {
    __syncthreads();
    // --- stage A tile 128x32 via async global->LDS (no VGPR round trip) ---
    const _Float16* ga = A + (size_t)(mbase + arow) * K + k0 + acol;
    async_copy_b128(ga,     &ldsA[arow][acol]);
    async_copy_b128(ga + 8, &ldsA[arow][acol + 8]);
    // --- stage B tile 32x64 transposed into ldsBt[n][k] (needs VGPR transpose) ---
    const _Float16* gb = Bm + (size_t)(k0 + krow) * N + nbase + ncol;
    v8h bv = *(const v8h*)gb;
#pragma unroll
    for (int i = 0; i < 8; ++i) ldsBt[ncol + i][krow] = bv[i];
    // prefetch next chunk into L2
    if (k0 + 32 < K) {
      __builtin_prefetch(ga + 32, 0, 1);
      __builtin_prefetch(gb + (size_t)32 * N, 0, 1);
    }
    wait_asynccnt0();
    __syncthreads();

    // A fragment: M=ln, K halves {kb..kb+7, kb+16..kb+23}
    v8h alo = *(const v8h*)&ldsA[w * 16 + ln][kb];
    v8h ahi = *(const v8h*)&ldsA[w * 16 + ln][kb + 16];
    v16h af = CAT16(alo, ahi);
#pragma unroll
    for (int nt = 0; nt < 4; ++nt) {
      v8h bl = *(const v8h*)&ldsBt[nt * 16 + ln][kb];
      v8h bh = *(const v8h*)&ldsBt[nt * 16 + ln][kb + 16];
      v16h bf = CAT16(bl, bh);
      acc[nt] = wmma_f16(af, bf, acc[nt]);
    }
  }

#pragma unroll
  for (int nt = 0; nt < 4; ++nt) {
#pragma unroll
    for (int j = 0; j < 8; ++j) {
      int rM = mbase + w * 16 + j + 8 * hl;      // C layout: VGPR j -> row j + 8*hl
      C[(size_t)rM * N + nbase + nt * 16 + ln] = acc[nt][j];
    }
  }
}

// ---------------- flash attention: one block = (b, h, 128 query rows) ----------------
__global__ __launch_bounds__(256) void attn_kernel(const _Float16* __restrict__ Q,
                                                   const _Float16* __restrict__ Kc,
                                                   const _Float16* __restrict__ Vc,
                                                   _Float16* __restrict__ O) {
  __shared__ alignas(16) _Float16 ldsK[32][136];     // [key][d], padded
  __shared__ alignas(16) _Float16 ldsVt[128][40];    // [d][key], padded (transposed)
  __shared__ alignas(16) _Float16 ldsP[8][16][40];   // per-wave P tile 16x32, padded

  const int bh = blockIdx.y;
  const int b  = bh / H_;
  const int h  = bh % H_;
  const int kv = h / (H_ / HKV_);
  const int qb = blockIdx.x * 128;

  const int tid  = threadIdx.x;
  const int lane = tid & 31;
  const int w    = tid >> 5;
  const int ln   = lane & 15;
  const int hl   = lane >> 4;
  const int kb   = hl ? 8 : 0;    // A-fragment K base
  const int kbb  = hl ? 16 : 0;   // unused; B fragments also use kb

  // Preload Q fragments for the 4 K-chunks over D (A layout: M=ln)
  const _Float16* qptr = Q + (((size_t)(b * T_ + qb + w * 16 + ln)) * H_ + h) * D_;
  v16h qfrag[4];
#pragma unroll
  for (int i = 0; i < 4; ++i) {
    v8h lo = *(const v8h*)(qptr + i * 32 + kb);
    v8h hi = *(const v8h*)(qptr + i * 32 + kb + 16);
    qfrag[i] = CAT16(lo, hi);
  }

  v8f acc[8];
#pragma unroll
  for (int i = 0; i < 8; ++i) acc[i] = (v8f){0,0,0,0,0,0,0,0};
  float mrow[8], lrow[8];
#pragma unroll
  for (int j = 0; j < 8; ++j) { mrow[j] = -1.0e30f; lrow[j] = 0.0f; }

  const float sc = 0.08838834764831845f;   // 1/sqrt(128)
  const int qmax = qb + w * 16 + 15;
  const int send = qb + 128;

  // staging coordinates: 32 keys x 128 d, 8 threads per key
  const int skey = tid >> 3, sdcol = (tid & 7) * 16;

  for (int s = 0; s < send; s += 32) {
    __syncthreads();
    // --- K tile: straight copy, async global->LDS ---
    const _Float16* kp = Kc + (((size_t)(b * T_ + s + skey)) * HKV_ + kv) * D_ + sdcol;
    async_copy_b128(kp,     &ldsK[skey][sdcol]);
    async_copy_b128(kp + 8, &ldsK[skey][sdcol + 8]);
    // --- V tile: transposed into ldsVt[d][key] via VGPRs ---
    const _Float16* vp = Vc + (((size_t)(b * T_ + s + skey)) * HKV_ + kv) * D_ + sdcol;
    v8h b0 = *(const v8h*)vp;
    v8h b1 = *(const v8h*)(vp + 8);
#pragma unroll
    for (int i = 0; i < 8; ++i) ldsVt[sdcol + i][skey]     = b0[i];
#pragma unroll
    for (int i = 0; i < 8; ++i) ldsVt[sdcol + 8 + i][skey] = b1[i];
    if (s + 32 < send) {
      __builtin_prefetch(kp + (size_t)32 * HKV_ * D_, 0, 1);
      __builtin_prefetch(vp + (size_t)32 * HKV_ * D_, 0, 1);
    }
    wait_asynccnt0();
    __syncthreads();
    if (s > qmax) continue;   // wave-uniform: EXEC stays all-ones for WMMA

    // ---- S = Q K^T for 32 keys (two 16x16 tiles) ----
    v8f s0 = (v8f){0,0,0,0,0,0,0,0};
    v8f s1 = (v8f){0,0,0,0,0,0,0,0};
#pragma unroll
    for (int i = 0; i < 4; ++i) {
      v8h k0l = *(const v8h*)&ldsK[ln][i * 32 + kb];
      v8h k0h = *(const v8h*)&ldsK[ln][i * 32 + kb + 16];
      v16h kf0 = CAT16(k0l, k0h);
      s0 = wmma_f16(qfrag[i], kf0, s0);
      v8h k1l = *(const v8h*)&ldsK[16 + ln][i * 32 + kb];
      v8h k1h = *(const v8h*)&ldsK[16 + ln][i * 32 + kb + 16];
      v16h kf1 = CAT16(k1l, k1h);
      s1 = wmma_f16(qfrag[i], kf1, s1);
    }

    // ---- online softmax (rows live per half-wave; butterfly over 16 lanes) ----
    float p0[8], p1[8], corr[8];
#pragma unroll
    for (int j = 0; j < 8; ++j) {
      int rq = qb + w * 16 + j + 8 * hl;
      float a  = s0[j] * sc; if (s + ln > rq)       a  = -3.0e38f;
      float bb = s1[j] * sc; if (s + 16 + ln > rq)  bb = -3.0e38f;
      float mx = fmaxf(a, bb);
      mx = fmaxf(mx, __shfl_xor(mx, 1));
      mx = fmaxf(mx, __shfl_xor(mx, 2));
      mx = fmaxf(mx, __shfl_xor(mx, 4));
      mx = fmaxf(mx, __shfl_xor(mx, 8));
      float mn = fmaxf(mrow[j], mx);
      float cr = __expf(mrow[j] - mn);
      float e0 = __expf(a - mn);
      float e1 = __expf(bb - mn);
      float sum = e0 + e1;
      sum += __shfl_xor(sum, 1);
      sum += __shfl_xor(sum, 2);
      sum += __shfl_xor(sum, 4);
      sum += __shfl_xor(sum, 8);
      lrow[j] = lrow[j] * cr + sum;
      mrow[j] = mn;
      corr[j] = cr;
      p0[j] = e0; p1[j] = e1;
    }
#pragma unroll
    for (int dt = 0; dt < 8; ++dt)
#pragma unroll
      for (int j = 0; j < 8; ++j) acc[dt][j] *= corr[j];

    // ---- P (C layout) -> LDS -> A-fragment layout ----
#pragma unroll
    for (int j = 0; j < 8; ++j) {
      ldsP[w][j + 8 * hl][ln]      = (_Float16)p0[j];
      ldsP[w][j + 8 * hl][16 + ln] = (_Float16)p1[j];
    }
    asm volatile("s_wait_dscnt 0" ::: "memory");
    v8h plo = *(const v8h*)&ldsP[w][ln][kb];
    v8h phi = *(const v8h*)&ldsP[w][ln][kb + 16];
    v16h pf = CAT16(plo, phi);

    // ---- ctx += P @ V over 8 d-tiles ----
#pragma unroll
    for (int dt = 0; dt < 8; ++dt) {
      v8h vl = *(const v8h*)&ldsVt[dt * 16 + ln][kb];
      v8h vh = *(const v8h*)&ldsVt[dt * 16 + ln][kb + 16];
      v16h vf = CAT16(vl, vh);
      acc[dt] = wmma_f16(pf, vf, acc[dt]);
    }
  }

  // ---- normalize and write ctx (f16) ----
#pragma unroll
  for (int dt = 0; dt < 8; ++dt) {
#pragma unroll
    for (int j = 0; j < 8; ++j) {
      int rq = qb + w * 16 + j + 8 * hl;
      O[(((size_t)(b * T_ + rq)) * H_ + h) * D_ + dt * 16 + ln] =
          (_Float16)(acc[dt][j] / lrow[j]);
    }
  }
}

// ---------------- host-side orchestration ----------------
extern "C" void kernel_launch(void* const* d_in, const int* in_sizes, int n_in,
                              void* d_out, int out_size, void* d_ws, size_t ws_size,
                              hipStream_t stream) {
  (void)in_sizes; (void)n_in; (void)out_size; (void)ws_size;
  const float* x    = (const float*)d_in[0];
  const float* Wq   = (const float*)d_in[1];
  const float* Wk   = (const float*)d_in[2];
  const float* Wv   = (const float*)d_in[3];
  const float* Wout = (const float*)d_in[4];

  char* ws = (char*)d_ws;
  size_t off = 0;
  auto alloc = [&](size_t bytes) { char* p = ws + off; off += (bytes + 255) & ~(size_t)255; return p; };

  _Float16* xh  = (_Float16*)alloc((size_t)BT_ * E_ * 2);
  _Float16* wqh = (_Float16*)alloc((size_t)E_ * H_ * D_ * 2);
  _Float16* wkh = (_Float16*)alloc((size_t)E_ * HKV_ * D_ * 2);
  _Float16* wvh = (_Float16*)alloc((size_t)E_ * HKV_ * D_ * 2);
  _Float16* woh = (_Float16*)alloc((size_t)H_ * D_ * E_ * 2);
  float*    qf  = (float*)   alloc((size_t)BT_ * H_ * D_ * 4);
  float*    kf  = (float*)   alloc((size_t)BT_ * HKV_ * D_ * 4);
  float*    vf  = (float*)   alloc((size_t)BT_ * HKV_ * D_ * 4);
  _Float16* qh  = (_Float16*)alloc((size_t)BT_ * H_ * D_ * 2);
  _Float16* kh  = (_Float16*)alloc((size_t)BT_ * HKV_ * D_ * 2);
  _Float16* vh  = (_Float16*)alloc((size_t)BT_ * HKV_ * D_ * 2);
  _Float16* ch  = (_Float16*)alloc((size_t)BT_ * H_ * D_ * 2);

  auto cvt = [&](const float* src, _Float16* dst, int n) {
    cvt_f32_f16_kernel<<<(n + 255) / 256, 256, 0, stream>>>(src, dst, n);
  };
  cvt(x,    xh,  BT_ * E_);
  cvt(Wq,   wqh, E_ * H_ * D_);
  cvt(Wk,   wkh, E_ * HKV_ * D_);
  cvt(Wv,   wvh, E_ * HKV_ * D_);
  cvt(Wout, woh, H_ * D_ * E_);

  // QKV projections (f32 out)
  gemm_f16_kernel<<<dim3((H_ * D_) / 64, BT_ / 128), 256, 0, stream>>>(xh, wqh, qf, BT_, H_ * D_, E_);
  gemm_f16_kernel<<<dim3((HKV_ * D_) / 64, BT_ / 128), 256, 0, stream>>>(xh, wkh, kf, BT_, HKV_ * D_, E_);
  gemm_f16_kernel<<<dim3((HKV_ * D_) / 64, BT_ / 128), 256, 0, stream>>>(xh, wvh, vf, BT_, HKV_ * D_, E_);

  // RoPE(q), RoPE(k), convert v
  {
    int nq = BT_ * H_ * (D_ / 2);
    rope_kernel<<<(nq + 255) / 256, 256, 0, stream>>>(qf, qh, H_, nq);
    int nk = BT_ * HKV_ * (D_ / 2);
    rope_kernel<<<(nk + 255) / 256, 256, 0, stream>>>(kf, kh, HKV_, nk);
    int nv = BT_ * HKV_ * D_;
    cvt_f32_f16_kernel<<<(nv + 255) / 256, 256, 0, stream>>>(vf, vh, nv);
  }

  // attention
  attn_kernel<<<dim3(T_ / 128, B_ * H_), 256, 0, stream>>>(qh, kh, vh, ch);

  // output projection -> d_out (f32)
  gemm_f16_kernel<<<dim3(E_ / 64, BT_ / 128), 256, 0, stream>>>(ch, woh, (float*)d_out, BT_, E_, H_ * D_);
}
